// Capsule_49555332661517
// MI455X (gfx1250) — matrix operand order repaired and checked by reading, not compiled, not run on hardware
//
#include <hip/hip_runtime.h>

typedef _Float16 half_t;
typedef _Float16 v16h __attribute__((ext_vector_type(16)));
typedef _Float16 v8h  __attribute__((ext_vector_type(8)));
typedef _Float16 v4h  __attribute__((ext_vector_type(4)));
typedef float    v8f  __attribute__((ext_vector_type(8)));

#define B_   64
#define I_   256
#define DIN  256
#define NC   32
#define DC   64
#define NO   2048   // NC*DC
#define LDA  40     // padded LDS row stride (halfs); 80B keeps 16B align, kills bank conflicts

// ---------------------------------------------------------------------------
// Stage 1: hat[b,n,i,d] = sum_k x[b,i,k] * kernel[i,k,n*64+d]
// One workgroup per (i, n). 128 threads = 4 waves.
// GEMM is computed TRANSPOSED: D[o][b] = kernelT[o][k] * xT[k][b]
//   A operand = kernel tile  (M = o, 16 rows per wave)
//   B operand = x tile       (N = b, 4 accumulator tiles of 16)
// so each lane's 8 accumulator entries are 8 CONSECUTIVE d values ->
// epilogue is 4x global_store_b128 per thread (d-contiguous in hat).
// ---------------------------------------------------------------------------
__global__ __launch_bounds__(128)
void hat_gemm(const float* __restrict__ x, const float* __restrict__ kern,
              half_t* __restrict__ hat)
{
    __shared__ half_t Ash[64 * LDA];   // x tile      [b][k]   (b=0..63, k=0..31)
    __shared__ half_t Bsh[64 * LDA];   // kernel tile [o][k]   (o=0..63 transposed)

    const int i    = blockIdx.x;
    const int n    = blockIdx.y;
    const int tid  = threadIdx.x;
    const int lane = tid & 31;
    const int w    = tid >> 5;               // wave id 0..3 -> o rows w*16..w*16+15
    const int ln15 = lane & 15;
    const int kbA  = (lane & 16) ? 8  : 0;   // A layout: hi lanes K=8..15 / 24..31
    const int kbB  = (lane & 16) ? 8  : 0;   // B layout: hi lanes K=8..15 / 24..31
    const int mhi  = (lane & 16) ? 8  : 0;   // C layout: hi lanes M=v+8

    v8f acc[4] = {};                         // acc[nt] : b-cols nt*16..nt*16+15

    const float* xi = x    + (size_t)i * DIN;                 // + b*I_*DIN + k
    const float* ki = kern + (size_t)i * DIN * NO + n * DC;   // + k*NO + o

    for (int k0 = 0; k0 < DIN; k0 += 32) {
        // --- stage x tile: 64 b-rows x 32 k (f32 -> f16), natural layout ---
        {
            const int kk    = (tid & 7) * 4;
            const int rbase = tid >> 3;            // 0..15
            #pragma unroll
            for (int r = 0; r < 4; ++r) {
                const int row = rbase + r * 16;
                const float4 v = *(const float4*)(xi + (size_t)row * (I_ * DIN) + k0 + kk);
                v4h h; h[0] = (half_t)v.x; h[1] = (half_t)v.y;
                       h[2] = (half_t)v.z; h[3] = (half_t)v.w;
                *(v4h*)(&Ash[row * LDA + kk]) = h;
            }
        }
        // --- stage kernel tile: 32 k x 64 o, transposed into Bsh[o][k] ---
        {
            const int cc  = (tid & 15) * 4;
            const int kk0 = tid >> 4;              // 0..7
            #pragma unroll
            for (int r = 0; r < 4; ++r) {
                const int k = kk0 + r * 8;
                const float4 v = *(const float4*)(ki + (size_t)(k0 + k) * NO + cc);
                Bsh[(cc + 0) * LDA + k] = (half_t)v.x;
                Bsh[(cc + 1) * LDA + k] = (half_t)v.y;
                Bsh[(cc + 2) * LDA + k] = (half_t)v.z;
                Bsh[(cc + 3) * LDA + k] = (half_t)v.w;
            }
        }
        __syncthreads();

        // A fragment (kernel): row o = w*16 + ln15, K = kbA..kbA+7, kbA+16..kbA+23
        v16h af;
        {
            const int o = w * 16 + ln15;
            const v8h alo = *(const v8h*)(&Bsh[o * LDA + kbA]);
            const v8h ahi = *(const v8h*)(&Bsh[o * LDA + kbA + 16]);
            #pragma unroll
            for (int t = 0; t < 8; ++t) { af[t] = alo[t]; af[8 + t] = ahi[t]; }
        }

        #pragma unroll
        for (int nt = 0; nt < 4; ++nt) {
            // B fragment (x): col b = nt*16 + ln15, K = kbB..kbB+7, kbB+16..kbB+23
            const int bcol = nt * 16 + ln15;
            const v8h blo = *(const v8h*)(&Ash[bcol * LDA + kbB]);
            const v8h bhi = *(const v8h*)(&Ash[bcol * LDA + kbB + 16]);
            v16h bf;
            #pragma unroll
            for (int t = 0; t < 8; ++t) { bf[t] = blo[t]; bf[8 + t] = bhi[t]; }
            acc[nt] = __builtin_amdgcn_wmma_f32_16x16x32_f16(
                          false, af, false, bf, (short)0, acc[nt], false, false);
        }
        __syncthreads();
    }

    // Epilogue: lane holds D[o = w*16 + mhi + v][b = nt*16 + ln15], v=0..7
    // -> 8 consecutive d per accumulator -> one b128 store each.
    const int d0 = w * 16 + mhi;
    #pragma unroll
    for (int nt = 0; nt < 4; ++nt) {
        const int b = nt * 16 + ln15;
        v8h h;
        #pragma unroll
        for (int v = 0; v < 8; ++v) h[v] = (half_t)acc[nt][v];
        *(v8h*)(hat + ((((size_t)b * NC + n) * I_ + i) * DC) + d0) = h;
    }
}

// ---------------------------------------------------------------------------
// Stage 2: dynamic routing (ROUTINGS = 2, unrolled). One block per batch b.
// ---------------------------------------------------------------------------
__global__ __launch_bounds__(256)
void routing(const half_t* __restrict__ hat, float* __restrict__ out)
{
    __shared__ float s1[NC * DC];      // 8 KB  (s1 / o1 / s2)
    __shared__ float c2s[NC * I_];     // 32 KB (routing coefficients)
    __shared__ float scale_s[NC];

    const int b   = blockIdx.x;
    const int tid = threadIdx.x;
    const half_t* H = hat + (size_t)b * NC * I_ * DC;   // [n][i][d]

    // r=0: softmax(0) = 1/32  ->  s1[n,d] = (1/32) * sum_i hat[n,i,d]
    #pragma unroll
    for (int c = 0; c < 8; ++c) {
        const int idx = tid + c * 256;
        const int n = idx >> 6, d = idx & 63;
        const half_t* p = H + (size_t)n * I_ * DC + d;
        float s = 0.f;
        #pragma unroll 8
        for (int i = 0; i < I_; ++i) s += (float)p[(size_t)i * DC];
        s1[idx] = s * (1.0f / 32.0f);
    }
    __syncthreads();
    if (tid < NC) {                                     // squash scale per n
        float sq = 1e-7f;
        for (int d = 0; d < DC; ++d) { const float v = s1[tid * DC + d]; sq += v * v; }
        scale_s[tid] = sqrtf(sq) / (1.0f + sq);
    }
    __syncthreads();
    #pragma unroll
    for (int c = 0; c < 8; ++c) { const int idx = tid + c * 256; s1[idx] *= scale_s[idx >> 6]; }
    __syncthreads();

    // agreement: b2[n, i=tid] = <o1[n,:], hat[n,tid,:]>, then softmax over n
    float b2[NC];
    for (int n = 0; n < NC; ++n) {
        const half_t* p = H + (size_t)n * I_ * DC + (size_t)tid * DC;
        float a = 0.f;
        #pragma unroll 8
        for (int d = 0; d < DC; ++d) a += s1[n * DC + d] * (float)p[d];
        b2[n] = a;
    }
    float mx = b2[0];
    #pragma unroll
    for (int n = 1; n < NC; ++n) mx = fmaxf(mx, b2[n]);
    float ssum = 0.f;
    #pragma unroll
    for (int n = 0; n < NC; ++n) { b2[n] = expf(b2[n] - mx); ssum += b2[n]; }
    const float inv = 1.0f / ssum;
    #pragma unroll
    for (int n = 0; n < NC; ++n) c2s[n * I_ + tid] = b2[n] * inv;
    __syncthreads();

    // r=1: s2[n,d] = sum_i c2[n,i] * hat[n,i,d], then squash and emit
    #pragma unroll
    for (int c = 0; c < 8; ++c) {
        const int idx = tid + c * 256;
        const int n = idx >> 6, d = idx & 63;
        const half_t* p = H + (size_t)n * I_ * DC + d;
        float s = 0.f;
        #pragma unroll 8
        for (int i = 0; i < I_; ++i) s += c2s[n * I_ + i] * (float)p[(size_t)i * DC];
        s1[idx] = s;
    }
    __syncthreads();
    if (tid < NC) {
        float sq = 1e-7f;
        for (int d = 0; d < DC; ++d) { const float v = s1[tid * DC + d]; sq += v * v; }
        scale_s[tid] = sqrtf(sq) / (1.0f + sq);
    }
    __syncthreads();
    #pragma unroll
    for (int c = 0; c < 8; ++c) {
        const int idx = tid + c * 256;
        out[(size_t)b * (NC * DC) + idx] = s1[idx] * scale_s[idx >> 6];
    }
}

extern "C" void kernel_launch(void* const* d_in, const int* in_sizes, int n_in,
                              void* d_out, int out_size, void* d_ws, size_t ws_size,
                              hipStream_t stream)
{
    (void)in_sizes; (void)n_in; (void)out_size; (void)ws_size;
    const float* x = (const float*)d_in[0];   // [64,256,256]
    const float* k = (const float*)d_in[1];   // [256,256,2048]
    half_t* hat = (half_t*)d_ws;              // needs 64*32*256*64*2 = 64 MiB

    hat_gemm<<<dim3(I_, NC), 128, 0, stream>>>(x, k, hat);
    routing<<<B_, 256, 0, stream>>>(hat, (float*)d_out);
}
